// MultiHeadAttention_10213432229911
// MI455X (gfx1250) — matrix operand order repaired and checked by reading, not compiled, not run on hardware
//
#include <hip/hip_runtime.h>
#include <stdint.h>

// ---------------------------------------------------------------------------
// Types for CDNA5 WMMA (wave32). D = A(16x32 bf16) * B(32x16 bf16) + C(16x16 f32)
// ---------------------------------------------------------------------------
typedef __attribute__((ext_vector_type(16))) __bf16        v16bf;
typedef __attribute__((ext_vector_type(8)))  float         v8f;
typedef __attribute__((ext_vector_type(4)))  unsigned int  v4u;
typedef __attribute__((ext_vector_type(8)))  int           v8i;
typedef __attribute__((ext_vector_type(4)))  int           v4i;

struct Frag {
    union { v16bf h; uint4 q[2]; };
};

__device__ __forceinline__ uint16_t f2bf(float f) {
    uint32_t u = __float_as_uint(f);
    uint32_t r = u + 0x7FFFu + ((u >> 16) & 1u);   // round to nearest even
    return (uint16_t)(r >> 16);
}

// A-matrix 16x32 bf16 fragment (ISA 7.12.2):
//   lanes 0-15 : M=lane,  v0..v3 = K[k0+0..7],  v4..v7 = K[k0+16..23]
//   lanes 16-31: M=lane-16, v0..v3 = K[k0+8..15], v4..v7 = K[k0+24..31]
template <typename PT>
__device__ __forceinline__ Frag load_a(const PT* base, int row, int ld, int k0, int lane) {
    const PT* p = base + row * ld + k0 + ((lane & 16) ? 8 : 0);
    Frag f;
    f.q[0] = *reinterpret_cast<const uint4*>(p);
    f.q[1] = *reinterpret_cast<const uint4*>(p + 16);
    return f;
}

// B-matrix 32x16 bf16 fragment (same K order as A):
//   lanes 0-15 : N=lane,  v0..v3 = K[k0+0..7],  v4..v7 = K[k0+16..23]
//   lanes 16-31: N=lane-16, v0..v3 = K[k0+8..15], v4..v7 = K[k0+24..31]
template <typename PT>
__device__ __forceinline__ Frag load_b(const PT* base, int row, int ld, int k0, int lane) {
    const PT* p = base + row * ld + k0 + ((lane & 16) ? 8 : 0);
    Frag f;
    f.q[0] = *reinterpret_cast<const uint4*>(p);
    f.q[1] = *reinterpret_cast<const uint4*>(p + 16);
    return f;
}

__device__ __forceinline__ v8f wmma_bf16(const Frag& a, const Frag& b, v8f c) {
    return __builtin_amdgcn_wmma_f32_16x16x32_bf16(
        /*neg_a=*/false, a.h, /*neg_b=*/false, b.h,
        /*c_mod=*/(short)0, c, /*reuse_a=*/false, /*reuse_b=*/false);
}

// ---------------------------------------------------------------------------
// TDM 2D tile load: Global -> LDS via Tensor Data Mover (ISA ch.8 descriptor).
// data_size = 2 bytes (bf16). group0: count=1, lds byte addr, 57b global addr,
// type=2. group1: data_size=1(2B), tensor_dim0/1, tile_dim0/1, dim0 stride.
// This toolchain's builtin is the 6-arg form:
//   (uint32x4 g0, int32x8 g1, int32x4 g2, int32x4 g3, int32x8 pad, i32 cpol)
// ---------------------------------------------------------------------------
__device__ __forceinline__ void tdm_load_2d(uint32_t lds_byte, uint64_t gaddr,
                                            uint32_t tensor_d0, uint32_t tensor_d1,
                                            uint32_t tile_d0, uint32_t tile_d1,
                                            uint32_t stride0) {
    v4u g0;
    g0.x = 1u;                                              // count=1, user mode
    g0.y = lds_byte;                                        // LDS byte address
    g0.z = (uint32_t)gaddr;                                 // global_addr[31:0]
    g0.w = (uint32_t)((gaddr >> 32) & 0x01FFFFFFu) | (2u << 30); // addr[56:32] | type=2
    v8i g1;
    g1[0] = (int)(1u << 16);                                // data_size=1 (2 bytes)
    g1[1] = (int)((tensor_d0 & 0xFFFFu) << 16);             // tensor_dim0[15:0]
    g1[2] = (int)((tensor_d0 >> 16) | ((tensor_d1 & 0xFFFFu) << 16));
    g1[3] = (int)((tensor_d1 >> 16) | (tile_d0 << 16));     // tensor_dim1 hi | tile_dim0
    g1[4] = (int)tile_d1;                                   // tile_dim1 (tile_dim2=0)
    g1[5] = (int)stride0;                                   // tensor_dim0_stride[31:0]
    g1[6] = 0;
    g1[7] = 0;
    v4i z4 = {0, 0, 0, 0};                                  // 2D: groups 2/3 unused
    v8i z8 = {0, 0, 0, 0, 0, 0, 0, 0};
    __builtin_amdgcn_tensor_load_to_lds(g0, g1, z4, z4, z8, 0);
}

// ---------------------------------------------------------------------------
// Problem constants
// ---------------------------------------------------------------------------
#define DM     1024
#define NH     16
#define HD     64
#define NB     2
#define SQ     1024
#define LC     1024
#define TT     2048         // LC + SQ
#define ROWS   (NB * SQ)    // 2048 token rows

// ---------------------------------------------------------------------------
// fp32 -> bf16 flat convert
// ---------------------------------------------------------------------------
__global__ void cvt_bf16_kernel(const float* __restrict__ src,
                                uint16_t* __restrict__ dst, int n) {
    int i = blockIdx.x * blockDim.x + threadIdx.x;
    if (i < n) dst[i] = f2bf(src[i]);
}

// cache_k [B,H,LC,HD] f32 -> Kbuf [B,H,TT,HD] bf16 (first LC rows of each slab)
__global__ void cache_k_kernel(const float* __restrict__ src,
                               uint16_t* __restrict__ dst) {
    int i = blockIdx.x * blockDim.x + threadIdx.x;     // over B*H*LC*HD
    int bh = i / (LC * HD);
    int w  = i - bh * (LC * HD);
    dst[(size_t)bh * TT * HD + w] = f2bf(src[i]);
}

// cache_v [B,H,LC,HD] f32 -> Vt [B,H,HD,TT] bf16 transposed (first LC cols)
__global__ void cache_v_kernel(const float* __restrict__ src,
                               uint16_t* __restrict__ dst) {
    int i = blockIdx.x * blockDim.x + threadIdx.x;     // over B*H*LC*HD
    int bh = i / (LC * HD);
    int w  = i - bh * (LC * HD);
    int t  = w / HD;
    int hd = w - t * HD;
    dst[((size_t)bh * HD + hd) * TT + t] = f2bf(src[i]);
}

// ---------------------------------------------------------------------------
// Projection GEMM, TDM-fed + double-buffered LDS staging:
//   C[m, e] = sum_d A[m,d] * W[e,d] + bias[e]
// Block = 256 threads (8 waves); block tile 128 x 64; wave tile 16 x 64.
// Per 64-wide k-step: wave0 DMAs A-tile (128x64), wave1 DMAs B-tile (64x64);
// all waves consume the previous buffer from LDS while the next DMA flies.
// mode 0: bf16 out [M, DM]                 (Q)
// mode 1: bf16 out -> Kbuf[b,h,LC+s,hd]    (K append)
// mode 2: bf16 out -> Vt  [b,h,hd,LC+s]    (V append, transposed)
// mode 3: f32  out [M, DM]                 (final projection)
// ---------------------------------------------------------------------------
#define KSTEP     64
#define A_ELEMS   (128 * KSTEP)                 // 8192 bf16 = 16 KB
#define B_ELEMS   (64 * KSTEP)                  // 4096 bf16 =  8 KB
#define GEMM_SMEM ((2 * A_ELEMS + 2 * B_ELEMS) * 2)   // 49152 bytes

__global__ __launch_bounds__(256)
void gemm_wmma_kernel(const uint16_t* __restrict__ A,
                      const uint16_t* __restrict__ W,
                      const float* __restrict__ bias,
                      void* __restrict__ out, int mode) {
    extern __shared__ uint16_t lds[];
    uint16_t* Abuf[2] = { lds,                lds + A_ELEMS };
    uint16_t* Bbuf[2] = { lds + 2 * A_ELEMS,  lds + 2 * A_ELEMS + B_ELEMS };
    const uint32_t A_off[2] = { 0u,                 (uint32_t)(A_ELEMS * 2) };
    const uint32_t B_off[2] = { (uint32_t)(2 * A_ELEMS * 2),
                                (uint32_t)(2 * A_ELEMS * 2 + B_ELEMS * 2) };

    const int lane = threadIdx.x & 31;
    const int wave = threadIdx.x >> 5;
    const int bn   = blockIdx.x & 15;          // 1024/64 = 16 col-blocks
    const int bm   = blockIdx.x >> 4;          // 2048/128 = 16 row-blocks
    const int m_blk = bm * 128;
    const int n_blk = bn * 64;
    const int m0   = m_blk + wave * 16;
    const int nlo  = lane & 15;

    const uint64_t Ag = (uint64_t)(uintptr_t)A;
    const uint64_t Wg = (uint64_t)(uintptr_t)W;

    auto issue = [&](int k0, int bufi) {
        if (wave == 0) {
            tdm_load_2d(A_off[bufi], Ag + ((uint64_t)m_blk * DM + k0) * 2,
                        /*tensor_d0=*/DM, /*tensor_d1=*/ROWS,
                        /*tile_d0=*/KSTEP, /*tile_d1=*/128, /*stride0=*/DM);
        } else if (wave == 1) {
            tdm_load_2d(B_off[bufi], Wg + ((uint64_t)n_blk * DM + k0) * 2,
                        /*tensor_d0=*/DM, /*tensor_d1=*/DM,
                        /*tile_d0=*/KSTEP, /*tile_d1=*/64, /*stride0=*/DM);
        }
    };

    v8f acc0 = {}, acc1 = {}, acc2 = {}, acc3 = {};

    issue(0, 0);
    for (int ks = 0; ks < DM / KSTEP; ++ks) {
        __builtin_amdgcn_s_wait_tensorcnt(0);       // issuing waves drain their DMA
        __syncthreads();                            // tile[ks&1] visible to all waves
        if (ks + 1 < DM / KSTEP) issue((ks + 1) * KSTEP, (ks + 1) & 1);

        const uint16_t* Ab = Abuf[ks & 1];
        const uint16_t* Bb = Bbuf[ks & 1];
#pragma unroll
        for (int kc = 0; kc < KSTEP; kc += 32) {
            Frag a  = load_a(Ab, wave * 16 + nlo, KSTEP, kc, lane);
            Frag b0 = load_b(Bb,  0 + nlo, KSTEP, kc, lane);
            Frag b1 = load_b(Bb, 16 + nlo, KSTEP, kc, lane);
            Frag b2 = load_b(Bb, 32 + nlo, KSTEP, kc, lane);
            Frag b3 = load_b(Bb, 48 + nlo, KSTEP, kc, lane);
            acc0 = wmma_bf16(a, b0, acc0);
            acc1 = wmma_bf16(a, b1, acc1);
            acc2 = wmma_bf16(a, b2, acc2);
            acc3 = wmma_bf16(a, b3, acc3);
        }
        __syncthreads();                            // reads done before refill
    }

    const int mofs = (lane & 16) ? 8 : 0;
    v8f accs[4] = {acc0, acc1, acc2, acc3};
#pragma unroll
    for (int j = 0; j < 4; ++j) {
        const int N  = n_blk + j * 16 + nlo;
        const float bv = bias[N];
#pragma unroll
        for (int r = 0; r < 8; ++r) {
            const int m = m0 + mofs + r;
            const float v = accs[j][r] + bv;
            if (mode == 3) {
                ((float*)out)[(size_t)m * DM + N] = v;
            } else {
                const uint16_t bf = f2bf(v);
                if (mode == 0) {
                    ((uint16_t*)out)[(size_t)m * DM + N] = bf;
                } else {
                    const int b = m >> 10, s = m & (SQ - 1);
                    const int h = N >> 6,  hd = N & (HD - 1);
                    const int bh = b * NH + h;
                    if (mode == 1)
                        ((uint16_t*)out)[((size_t)bh * TT + LC + s) * HD + hd] = bf;
                    else  // mode 2: transposed V
                        ((uint16_t*)out)[((size_t)bh * HD + hd) * TT + LC + s] = bf;
                }
            }
        }
    }
}

// ---------------------------------------------------------------------------
// Attention core: one workgroup (128 thr = 4 waves) per (b, h, 16-query tile).
// Whole 16 x 2048 score block lives in LDS (fp32, 128KB) + bf16 P (64KB):
// CDNA5's 320KB WGP LDS makes the simple two-pass softmax viable.
//   Qb : [ROWS, DM] bf16 (projection layout)
//   Kb : [B,H,TT,HD] bf16
//   Vt : [B,H,HD,TT] bf16
//   Ob : [ROWS, DM] bf16
// ---------------------------------------------------------------------------
#define SMEM_SC   (16 * TT * 4)                 // 131072
#define SMEM_P    (16 * TT * 2)                 // 65536
#define SMEM_RED  (16 * 8 * 4)                  // 512
#define SMEM_STAT (16 * 4)                      // 64
#define SMEM_TOTAL (SMEM_SC + SMEM_P + SMEM_RED + SMEM_STAT)

__global__ __launch_bounds__(128)
void attn_wmma_kernel(const uint16_t* __restrict__ Qb,
                      const uint16_t* __restrict__ Kb,
                      const uint16_t* __restrict__ Vt,
                      uint16_t* __restrict__ Ob) {
    extern __shared__ char smem[];
    float*    sc      = (float*)smem;                         // [16][TT]
    uint16_t* P       = (uint16_t*)(smem + SMEM_SC);          // [16][TT]
    float*    red     = (float*)(smem + SMEM_SC + SMEM_P);    // [16][8]
    float*    rowstat = (float*)(smem + SMEM_SC + SMEM_P + SMEM_RED); // [16]

    const int tid  = threadIdx.x;
    const int lane = tid & 31;
    const int wave = tid >> 5;

    const int qt = blockIdx.x & 63;            // SQ/16 query tiles
    const int bh = blockIdx.x >> 6;            // b*NH + h
    const int b  = bh >> 4, h = bh & 15;
    const int q0 = qt * 16;

    const uint16_t* Kbh = Kb + (size_t)bh * TT * HD;
    const uint16_t* Vbh = Vt + (size_t)bh * HD * TT;
    const int qrow = b * SQ + q0;              // row base in [ROWS, DM]
    const int qcol = h * HD;

    // Q fragments: 16 queries x 64(hd) -> two K=32 chunks
    Frag qa0 = load_a(Qb, qrow + (lane & 15), DM, qcol +  0, lane);
    Frag qa1 = load_a(Qb, qrow + (lane & 15), DM, qcol + 32, lane);

    const float scale  = 0.125f;               // 1/sqrt(64)
    const int   offset = TT - SQ;              // causal cache offset = 1024
    const float NEGINF = -__builtin_inff();

    // ---- Phase A: scores = Q K^T, scaled + causal-masked, into LDS ----
    for (int kt = wave; kt < TT / 16; kt += 4) {
        const int t0 = kt * 16;
        Frag kb0 = load_b(Kbh, t0 + (lane & 15), HD,  0, lane);
        Frag kb1 = load_b(Kbh, t0 + (lane & 15), HD, 32, lane);
        v8f c = {};
        c = wmma_bf16(qa0, kb0, c);
        c = wmma_bf16(qa1, kb1, c);
        const int n  = lane & 15;
        const int mo = (lane & 16) ? 8 : 0;
        const int t  = t0 + n;
#pragma unroll
        for (int r = 0; r < 8; ++r) {
            const int q = q0 + mo + r;
            sc[(mo + r) * TT + t] = (t > q + offset) ? NEGINF : c[r] * scale;
        }
    }
    __syncthreads();

    // ---- Phase B: two-pass softmax over each of 16 rows of length 2048 ----
    const int rrow  = tid & 15;
    const int chunk = tid >> 4;                // 8 chunks x 256 cols
    {
        const float* p = sc + rrow * TT + chunk * 256;
        float m = NEGINF;
        for (int i = 0; i < 256; ++i) m = fmaxf(m, p[i]);
        red[rrow * 8 + chunk] = m;
    }
    __syncthreads();
    if (tid < 16) {
        float m = red[tid * 8];
        for (int i = 1; i < 8; ++i) m = fmaxf(m, red[tid * 8 + i]);
        rowstat[tid] = m;
    }
    __syncthreads();
    {
        const float m = rowstat[rrow];
        float*    p  = sc + rrow * TT + chunk * 256;
        uint16_t* pp = P  + rrow * TT + chunk * 256;
        float s = 0.f;
        for (int i = 0; i < 256; ++i) {
            const float e = __expf(p[i] - m);
            s += e;
            pp[i] = f2bf(e);
        }
        red[rrow * 8 + chunk] = s;
    }
    __syncthreads();
    if (tid < 16) {
        float s = 0.f;
        for (int i = 0; i < 8; ++i) s += red[tid * 8 + i];
        rowstat[tid] = 1.0f / s;
    }
    __syncthreads();

    // ---- Phase C: O = P @ V ; wave owns one 16-wide hd slice ----
    const int n0 = wave * 16;
    v8f o = {};
    for (int k0 = 0; k0 < TT; k0 += 32) {
        Frag a  = load_a(P,   lane & 15,        TT, k0, lane);   // ds_load_b128
        Frag bb = load_b(Vbh, n0 + (lane & 15), TT, k0, lane);
        o = wmma_bf16(a, bb, o);
    }
    const int n  = lane & 15;
    const int mo = (lane & 16) ? 8 : 0;
#pragma unroll
    for (int r = 0; r < 8; ++r) {
        const int m = mo + r;
        Ob[(size_t)(qrow + m) * DM + qcol + n0 + n] = f2bf(o[r] * rowstat[m]);
    }
}

// ---------------------------------------------------------------------------
// Host-side orchestration
// ---------------------------------------------------------------------------
extern "C" void kernel_launch(void* const* d_in, const int* in_sizes, int n_in,
                              void* d_out, int out_size, void* d_ws, size_t ws_size,
                              hipStream_t stream) {
    const float* x  = (const float*)d_in[0];
    const float* ck = (const float*)d_in[1];
    const float* cv = (const float*)d_in[2];
    const float* Wq = (const float*)d_in[3];
    const float* bq = (const float*)d_in[4];
    const float* Wk = (const float*)d_in[5];
    const float* bk = (const float*)d_in[6];
    const float* Wv = (const float*)d_in[7];
    const float* bv = (const float*)d_in[8];
    const float* Wo = (const float*)d_in[9];
    const float* bo = (const float*)d_in[10];

    // Workspace layout (bytes)
    char* ws = (char*)d_ws;
    const size_t SZ_X = (size_t)ROWS * DM * 2;         // 4 MB
    const size_t SZ_W = (size_t)DM * DM * 2;           // 2 MB
    const size_t SZ_K = (size_t)NB * NH * TT * HD * 2; // 8 MB
    uint16_t* xb  = (uint16_t*)(ws);
    uint16_t* Wqb = (uint16_t*)(ws + SZ_X);
    uint16_t* Wkb = (uint16_t*)(ws + SZ_X + SZ_W);
    uint16_t* Wvb = (uint16_t*)(ws + SZ_X + 2 * SZ_W);
    uint16_t* Wob = (uint16_t*)(ws + SZ_X + 3 * SZ_W);
    uint16_t* Qb  = (uint16_t*)(ws + SZ_X + 4 * SZ_W);
    uint16_t* Kbuf= (uint16_t*)(ws + 2 * SZ_X + 4 * SZ_W);
    uint16_t* Vt  = (uint16_t*)(ws + 2 * SZ_X + 4 * SZ_W + SZ_K);
    uint16_t* Ob  = (uint16_t*)(ws + 2 * SZ_X + 4 * SZ_W + 2 * SZ_K);

    const int nx = ROWS * DM;         // 2M
    const int nw = DM * DM;           // 1M
    const int nc = NB * NH * LC * HD; // 2M

    cvt_bf16_kernel<<<nx / 256, 256, 0, stream>>>(x,  xb,  nx);
    cvt_bf16_kernel<<<nw / 256, 256, 0, stream>>>(Wq, Wqb, nw);
    cvt_bf16_kernel<<<nw / 256, 256, 0, stream>>>(Wk, Wkb, nw);
    cvt_bf16_kernel<<<nw / 256, 256, 0, stream>>>(Wv, Wvb, nw);
    cvt_bf16_kernel<<<nw / 256, 256, 0, stream>>>(Wo, Wob, nw);
    cache_k_kernel<<<nc / 256, 256, 0, stream>>>(ck, Kbuf);
    cache_v_kernel<<<nc / 256, 256, 0, stream>>>(cv, Vt);

    // 256 blocks = (2048/128) row-blocks x (1024/64) col-blocks, 48KB LDS each
    gemm_wmma_kernel<<<256, 256, GEMM_SMEM, stream>>>(xb, Wqb, bq, Qb,   0);
    gemm_wmma_kernel<<<256, 256, GEMM_SMEM, stream>>>(xb, Wkb, bk, Kbuf, 1);
    gemm_wmma_kernel<<<256, 256, GEMM_SMEM, stream>>>(xb, Wvb, bv, Vt,   2);

    // one block per (b, h, 16-query tile): 2*16*64 = 2048 blocks, 197KB LDS
    attn_wmma_kernel<<<NB * NH * (SQ / 16), 128, SMEM_TOTAL, stream>>>(Qb, Kbuf, Vt, Ob);

    gemm_wmma_kernel<<<256, 256, GEMM_SMEM, stream>>>(Ob, Wob, bo, d_out, 3);
}
